// GAT_5317169512696
// MI455X (gfx1250) — matrix operand order repaired and checked by reading, not compiled, not run on hardware
//
#include <hip/hip_runtime.h>
#include <hip/hip_bf16.h>
#include <stdint.h>
#include <stddef.h>

// ---------------------------------------------------------------------------
// GAT (3x GATConv heads=2 + MLP head) for MI455X (gfx1250, wave32, WMMA).
// GEMMs: v_wmma_f32_16x16x32_bf16. Weight slab staged into LDS by the Tensor
// Data Mover (tensor_load_to_lds + s_wait_tensorcnt); A stream software-
// pipelined 2-deep in registers; B fragments batch-loaded from LDS.
// Message passing uses f32 L2 atomics; segment-max via ordered-uint atomicMax.
// ---------------------------------------------------------------------------

typedef __bf16 bf16;
typedef bf16  bf16x8  __attribute__((ext_vector_type(8)));
typedef bf16  bf16x16 __attribute__((ext_vector_type(16)));
typedef float f32x8   __attribute__((ext_vector_type(8)));
typedef unsigned int v4u __attribute__((ext_vector_type(4)));
typedef int          v8i __attribute__((ext_vector_type(8)));
typedef int          v4i __attribute__((ext_vector_type(4)));

#define NEG_SLOPE 0.2f
#define IN_DIM 768
#define KSLAB 256   // K-slab staged in LDS per round (all layer K's are multiples)

// ---- float <-> orderable-uint (for atomic float max) ----------------------
__device__ __forceinline__ unsigned f2o(float f) {
    unsigned u = __float_as_uint(f);
    return (u & 0x80000000u) ? ~u : (u | 0x80000000u);
}
__device__ __forceinline__ float o2f(unsigned o) {
    unsigned u = (o & 0x80000000u) ? (o & 0x7FFFFFFFu) : ~o;
    return __uint_as_float(u);
}

// ---------------------------------------------------------------------------
// TDM: DMA a 2D tile (rows x rowBytes, row stride strideBytes) from global
// memory into LDS at ldsOff. Issued by one wave; caller syncs the workgroup.
// D# layout per CDNA5 ISA 08_async_tensor §8.3/§8.4 (data_size = 1 byte).
// ---------------------------------------------------------------------------
__device__ __forceinline__ void tdm_load_2d(unsigned ldsOff, const void* gsrc,
                                            unsigned rows, unsigned rowBytes,
                                            unsigned strideBytes) {
    const unsigned long long ga = (unsigned long long)(uintptr_t)gsrc;
    v4u g0;
    g0[0] = 1u;                                  // count=1 (valid), user mode
    g0[1] = ldsOff;                              // [63:32]  lds_addr
    g0[2] = (unsigned)(ga & 0xFFFFFFFFu);        // [95:64]  global_addr lo
    g0[3] = (unsigned)((ga >> 32) & 0x01FFFFFFu) // [120:96] global_addr hi
          | (2u << 30);                          // [127:126] type = 2 (image)
    v8i g1;
    g1[0] = 0;                                   // wg_mask=0, data_size=1B, no flags
    g1[1] = (int)((rowBytes & 0xFFFFu) << 16);   // [63:48]   tensor_dim0 lo16
    g1[2] = (int)(((rowBytes >> 16) & 0xFFFFu)   // [79:64]   tensor_dim0 hi16
          | ((rows & 0xFFFFu) << 16));           // [95:80]   tensor_dim1 lo16
    g1[3] = (int)(((rows >> 16) & 0xFFFFu)       // [111:96]  tensor_dim1 hi16
          | ((rowBytes & 0xFFFFu) << 16));       // [127:112] tile_dim0 = rowBytes
    g1[4] = (int)(rows & 0xFFFFu);               // [143:128] tile_dim1 = rows
    g1[5] = (int)strideBytes;                    // [191:160] tensor_dim0_stride lo32
    g1[6] = 0;                                   // stride hi16, dim1_stride lo16
    g1[7] = 0;
    v4i z4 = {0, 0, 0, 0};
    v8i z8 = {0, 0, 0, 0, 0, 0, 0, 0};
    __builtin_amdgcn_tensor_load_to_lds(g0, g1, z4, z4, z8, 0);
}

// ---------------------------------------------------------------------------
// WMMA GEMM: C[M,N] = A[M,K] * Wt[N,K]^T  (A row-major bf16; Wt = W transposed,
// row-major bf16 so both fragments load contiguously).
// Block = 256 threads = 8 waves. Each wave: one 16-row M tile x 64-wide N slab
// (4 f32x8 accumulators). B slab DMA'd to LDS by TDM; A pipelined from global.
//
// Fragment layouts loaded here (wave32; contiguous k, same order for A and B):
//  A 16x32 bf16: lane<16 -> row = lane,    K = {k..k+15}
//                lane>=16 -> row = lane-16, K = {k+16..k+31}
//  B 32x16 bf16: lane<16 -> col = lane,    K = {k..k+15}
//                lane>=16 -> col = lane-16, K = {k+16..k+31}
// ---------------------------------------------------------------------------
__global__ void __launch_bounds__(256) gemm_bf16_wmma(
        const bf16* __restrict__ A, const bf16* __restrict__ Wt,
        int M, int K, int N,
        bf16* __restrict__ outB, float* __restrict__ outF,
        const float* __restrict__ bias, int doElu)
{
    __shared__ bf16 sB[64 * KSLAB];   // 32 KB: 64 Wt rows x KSLAB columns

    const int tid  = threadIdx.x;
    const int lane = tid & 31;
    const int wave = tid >> 5;
    const int r16  = lane & 15;
    const int kh   = lane >> 4;       // lane half
    const int n0   = blockIdx.y * 64;

    const int mtiles = M / 16;        // M divisible by 16
    int mtile = blockIdx.x * 8 + wave;
    const bool active = (mtile < mtiles);
    if (!active) mtile = mtiles - 1;  // clamp so loads stay in-bounds
    const int m0 = mtile * 16;

    const bf16* arow = A + (size_t)(m0 + r16) * K + kh * 16;

    auto loadA = [&](int k) -> bf16x16 {
        bf16x8 lo = *(const bf16x8*)(arow + k);
        bf16x8 hi = *(const bf16x8*)(arow + k + 8);
        return __builtin_shufflevector(lo, hi,
            0,1,2,3,4,5,6,7,8,9,10,11,12,13,14,15);
    };

    f32x8 acc[4] = {};
    // 2-deep A pipeline (K >= 256 always here).
    bf16x16 aCur = loadA(0);
    bf16x16 aNxt = loadA(32);

    const int sbase = r16 * KSLAB + kh * 16;          // per-lane LDS fragment base
    const unsigned sBoff = (unsigned)(uintptr_t)sB;   // LDS byte offset of sB

    for (int ks = 0; ks < K; ks += KSLAB) {
        __syncthreads();   // all waves done reading previous slab
        if (wave == 0) {
            // DMA Wt[n0..n0+64) x [ks..ks+KSLAB) into LDS (one TDM op).
            tdm_load_2d(sBoff, Wt + (size_t)n0 * K + ks,
                        64u, (unsigned)(KSLAB * 2), (unsigned)(K * 2));
            __builtin_amdgcn_s_wait_tensorcnt(0);
        }
        __syncthreads();

        for (int kk = 0; kk < KSLAB; kk += 32) {
            const int k = ks + kk;
            // Batch all four B fragments so the ds_load waits can stagger.
            bf16x16 bfr[4];
#pragma unroll
            for (int t = 0; t < 4; ++t) {
                const bf16* bp = sB + (sbase + t * 16 * KSLAB + kk);
                bf16x8 blo = *(const bf16x8*)bp;
                bf16x8 bhi = *(const bf16x8*)(bp + 8);
                bfr[t] = __builtin_shufflevector(blo, bhi,
                    0,1,2,3,4,5,6,7,8,9,10,11,12,13,14,15);
            }
            bf16x16 aFut = aCur;
            if (k + 64 < K) aFut = loadA(k + 64);   // prefetch 2 steps ahead
#pragma unroll
            for (int t = 0; t < 4; ++t) {
                acc[t] = __builtin_amdgcn_wmma_f32_16x16x32_bf16(
                    false, aCur, false, bfr[t], (short)0, acc[t], false, false);
            }
            aCur = aNxt;
            aNxt = aFut;
        }
    }

    if (!active) return;

    // D layout: VGPR r, lanes 0-15 -> (M = r,   N = lane)
    //                   lanes16-31 -> (M = 8+r, N = lane-16)
#pragma unroll
    for (int t = 0; t < 4; ++t) {
        const int n = n0 + t * 16 + r16;
        const float bval = bias ? bias[n] : 0.0f;
#pragma unroll
        for (int rr = 0; rr < 8; ++rr) {
            const int m = m0 + kh * 8 + rr;
            float v = acc[t][rr] + bval;
            if (doElu) v = (v > 0.0f) ? v : (expf(v) - 1.0f);
            if (outF) outF[(size_t)m * N + n] = v;
            if (outB) outB[(size_t)m * N + n] = (bf16)v;
        }
    }
}

// ---- helpers --------------------------------------------------------------
__global__ void f32_to_bf16_kernel(const float* __restrict__ src,
                                   bf16* __restrict__ dst, size_t n) {
    size_t i = (size_t)blockIdx.x * blockDim.x + threadIdx.x;
    if (i < n) dst[i] = (bf16)src[i];
}

// Convert + transpose weights: W[K,N] f32 -> Wt[N,K] bf16.
__global__ void wt_bf16_kernel(const float* __restrict__ W,
                               bf16* __restrict__ Wt, int K, int N) {
    int i = blockIdx.x * blockDim.x + threadIdx.x;
    if (i >= K * N) return;
    int k = i / N, n = i % N;
    Wt[(size_t)n * K + k] = (bf16)W[i];
}

__global__ void fill_u32_kernel(unsigned* __restrict__ p, unsigned v, int n) {
    int i = blockIdx.x * blockDim.x + threadIdx.x;
    if (i < n) p[i] = v;
}

// Per-(node, head) attention logits from pre-activation h (bf16).
__global__ void attn_logits_kernel(const bf16* __restrict__ H,
                                   const float* __restrict__ asrc,
                                   const float* __restrict__ adst,
                                   float* __restrict__ als,
                                   float* __restrict__ ald,
                                   int Nn, int C) {
    int idx = blockIdx.x * blockDim.x + threadIdx.x;
    if (idx >= Nn * 2) return;
    int n = idx >> 1, h = idx & 1;
    const bf16*  row = H + (size_t)n * (2 * C) + (size_t)h * C;
    const float* vs  = asrc + (size_t)h * C;
    const float* vd  = adst + (size_t)h * C;
    float s = 0.0f, d = 0.0f;
    for (int c = 0; c < C; ++c) {
        float v = (float)row[c];
        s = fmaf(v, vs[c], s);
        d = fmaf(v, vd[c], d);
    }
    als[idx] = s;
    ald[idx] = d;
}

__device__ __forceinline__ void edge_sd(int e, int Ereal,
                                        const int* esrc, const int* edst,
                                        int& s, int& d) {
    if (e < Ereal) { s = esrc[e]; d = edst[e]; }
    else           { s = e - Ereal; d = s; }   // self loops appended
}

// Segment max over incoming edges (ordered-uint atomicMax).
__global__ void edge_max_kernel(const int* __restrict__ esrc,
                                const int* __restrict__ edst,
                                int Ereal, int Nn,
                                const float* __restrict__ als,
                                const float* __restrict__ ald,
                                unsigned* __restrict__ mx) {
    int e = blockIdx.x * blockDim.x + threadIdx.x;
    if (e >= Ereal + Nn) return;
    int s, d; edge_sd(e, Ereal, esrc, edst, s, d);
#pragma unroll
    for (int h = 0; h < 2; ++h) {
        float v = als[s * 2 + h] + ald[d * 2 + h];
        v = (v > 0.0f) ? v : NEG_SLOPE * v;      // leaky_relu
        atomicMax(&mx[d * 2 + h], f2o(v));
    }
}

// w = exp(lr - max); accumulate denominator per destination.
__global__ void edge_exp_kernel(const int* __restrict__ esrc,
                                const int* __restrict__ edst,
                                int Ereal, int Nn,
                                const float* __restrict__ als,
                                const float* __restrict__ ald,
                                const unsigned* __restrict__ mx,
                                float* __restrict__ den,
                                float* __restrict__ ew) {
    int e = blockIdx.x * blockDim.x + threadIdx.x;
    if (e >= Ereal + Nn) return;
    int s, d; edge_sd(e, Ereal, esrc, edst, s, d);
#pragma unroll
    for (int h = 0; h < 2; ++h) {
        float v = als[s * 2 + h] + ald[d * 2 + h];
        v = (v > 0.0f) ? v : NEG_SLOPE * v;
        float w = expf(v - o2f(mx[d * 2 + h]));
        ew[(size_t)e * 2 + h] = w;
        atomicAdd(&den[d * 2 + h], w);
    }
}

// alpha = w / denom[dst]
__global__ void edge_alpha_kernel(const int* __restrict__ edst,
                                  int Ereal, int Nn,
                                  const float* __restrict__ den,
                                  float* __restrict__ ew) {
    int e = blockIdx.x * blockDim.x + threadIdx.x;
    if (e >= Ereal + Nn) return;
    int d = (e < Ereal) ? edst[e] : (e - Ereal);
#pragma unroll
    for (int h = 0; h < 2; ++h)
        ew[(size_t)e * 2 + h] /= den[d * 2 + h];
}

// Scatter h[src] * alpha into AGG[dst] (one block per edge, f32 L2 atomics).
__global__ void aggregate_kernel(const int* __restrict__ esrc,
                                 const int* __restrict__ edst,
                                 int Ereal,
                                 const bf16* __restrict__ Hs,
                                 const float* __restrict__ alpha,
                                 float* __restrict__ AGG, int C) {
    int e = blockIdx.x;
    int s, d; edge_sd(e, Ereal, esrc, edst, s, d);
    const int HC = 2 * C;
    const bf16* hrow = Hs  + (size_t)s * HC;
    float*      arow = AGG + (size_t)d * HC;
    const float a0 = alpha[(size_t)e * 2 + 0];
    const float a1 = alpha[(size_t)e * 2 + 1];
    for (int j = threadIdx.x; j < HC; j += blockDim.x) {
        float av = (j < C) ? a0 : a1;
        atomicAdd(&arow[j], (float)hrow[j] * av);
    }
}

// out = bf16(elu(AGG + bias)) -> input buffer of next layer.
__global__ void finish_kernel(const float* __restrict__ AGG,
                              const float* __restrict__ bias,
                              bf16* __restrict__ out,
                              size_t total, int HC) {
    size_t i = (size_t)blockIdx.x * blockDim.x + threadIdx.x;
    if (i >= total) return;
    int j = (int)(i % (size_t)HC);
    float v = AGG[i] + bias[j];
    v = (v > 0.0f) ? v : (expf(v) - 1.0f);
    out[i] = (bf16)v;
}

// 256 -> 5 classifier head (scalar fp32; trivial FLOPs).
__global__ void head_kernel(const float* __restrict__ hf,
                            const float* __restrict__ W5,
                            const float* __restrict__ b5,
                            float* __restrict__ out, int Nn) {
    int i = blockIdx.x * blockDim.x + threadIdx.x;
    if (i >= Nn * 5) return;
    int n = i / 5, c = i % 5;
    const float* row = hf + (size_t)n * 256;
    float acc = b5[c];
    for (int k = 0; k < 256; ++k)
        acc = fmaf(row[k], W5[k * 5 + c], acc);
    out[i] = acc;
}

// ---------------------------------------------------------------------------
extern "C" void kernel_launch(void* const* d_in, const int* in_sizes, int n_in,
                              void* d_out, int out_size, void* d_ws, size_t ws_size,
                              hipStream_t stream)
{
    (void)n_in; (void)out_size; (void)ws_size;

    const float* x   = (const float*)d_in[0];
    const int*   ei  = (const int*)d_in[1];
    const int M      = in_sizes[0] / IN_DIM;     // 50000
    const int Ereal  = in_sizes[1] / 2;          // 400000
    const int E      = Ereal + M;                // + self loops
    const int* esrc  = ei;
    const int* edst  = ei + Ereal;

    // params: W1,as1,ad1,b1, W2,as2,ad2,b2, W3,as3,ad3,b3, W4,b4, W5,b5
    const float* Wl[3] = {(const float*)d_in[2],  (const float*)d_in[6],  (const float*)d_in[10]};
    const float* As[3] = {(const float*)d_in[3],  (const float*)d_in[7],  (const float*)d_in[11]};
    const float* Ad[3] = {(const float*)d_in[4],  (const float*)d_in[8],  (const float*)d_in[12]};
    const float* Bb[3] = {(const float*)d_in[5],  (const float*)d_in[9],  (const float*)d_in[13]};
    const float* W4 = (const float*)d_in[14];
    const float* b4 = (const float*)d_in[15];
    const float* W5 = (const float*)d_in[16];
    const float* b5 = (const float*)d_in[17];
    const int Kdim[3] = {768, 768, 512};
    const int Cdim[3] = {384, 256, 256};

    // workspace carve-out (sized at max layer dims)
    char* ws = (char*)d_ws;
    auto aln = [](size_t v) { return (v + 255) & ~(size_t)255; };
    size_t off = 0;
    bf16*  Abuf = (bf16*)(ws + off);  off = aln(off + (size_t)M * 768 * 2);   // layer input (bf16)
    bf16*  Hbuf = (bf16*)(ws + off);  off = aln(off + (size_t)M * 768 * 2);   // pre-agg h (bf16)
    float* AGG  = (float*)(ws + off); off = aln(off + (size_t)M * 768 * 4);   // f32 aggregation
    bf16*  Wt   = (bf16*)(ws + off);  off = aln(off + (size_t)768 * 768 * 2); // transposed bf16 W
    float* ALS  = (float*)(ws + off); off = aln(off + (size_t)M * 2 * 4);
    float* ALD  = (float*)(ws + off); off = aln(off + (size_t)M * 2 * 4);
    unsigned* MX = (unsigned*)(ws + off); off = aln(off + (size_t)M * 2 * 4);
    float* DEN  = (float*)(ws + off); off = aln(off + (size_t)M * 2 * 4);
    float* EW   = (float*)(ws + off); off = aln(off + (size_t)E * 2 * 4);

    float* hout   = (float*)d_out;              // [M, 256]
    float* logits = hout + (size_t)M * 256;     // [M, 5]

    const int TB = 256;
    const int mblocks = (((M + 15) / 16) + 7) / 8;

    // x -> bf16
    {
        size_t n = (size_t)M * IN_DIM;
        f32_to_bf16_kernel<<<(unsigned)((n + TB - 1) / TB), TB, 0, stream>>>(x, Abuf, n);
    }

    for (int L = 0; L < 3; ++L) {
        const int K = Kdim[L], C = Cdim[L], HC = 2 * C;

        wt_bf16_kernel<<<(K * HC + TB - 1) / TB, TB, 0, stream>>>(Wl[L], Wt, K, HC);

        gemm_bf16_wmma<<<dim3(mblocks, HC / 64), TB, 0, stream>>>(
            Abuf, Wt, M, K, HC, Hbuf, nullptr, nullptr, 0);

        (void)hipMemsetAsync(AGG, 0, (size_t)M * HC * 4, stream);
        (void)hipMemsetAsync(DEN, 0, (size_t)M * 2 * 4, stream);
        fill_u32_kernel<<<(M * 2 + TB - 1) / TB, TB, 0, stream>>>(MX, 0x007FFFFFu /* f2o(-inf) */, M * 2);

        attn_logits_kernel<<<(M * 2 + TB - 1) / TB, TB, 0, stream>>>(
            Hbuf, As[L], Ad[L], ALS, ALD, M, C);

        edge_max_kernel<<<(E + TB - 1) / TB, TB, 0, stream>>>(
            esrc, edst, Ereal, M, ALS, ALD, MX);
        edge_exp_kernel<<<(E + TB - 1) / TB, TB, 0, stream>>>(
            esrc, edst, Ereal, M, ALS, ALD, MX, DEN, EW);
        edge_alpha_kernel<<<(E + TB - 1) / TB, TB, 0, stream>>>(
            edst, Ereal, M, DEN, EW);
        aggregate_kernel<<<E, TB, 0, stream>>>(
            esrc, edst, Ereal, Hbuf, EW, AGG, C);

        size_t total = (size_t)M * HC;
        finish_kernel<<<(unsigned)((total + TB - 1) / TB), TB, 0, stream>>>(
            AGG, Bb[L], Abuf, total, HC);
    }

    // Layer 4: h = elu(A @ W4 + b4), f32 straight into d_out.
    wt_bf16_kernel<<<(512 * 256 + TB - 1) / TB, TB, 0, stream>>>(W4, Wt, 512, 256);
    gemm_bf16_wmma<<<dim3(mblocks, 256 / 64), TB, 0, stream>>>(
        Abuf, Wt, M, 512, 256, nullptr, hout, b4, 1);

    // Layer 5: logits.
    head_kernel<<<(M * 5 + TB - 1) / TB, TB, 0, stream>>>(hout, W5, b5, logits, M);
}
